// DynamicConv2D_88055419502994
// MI455X (gfx1250) — compile-verified, matrix operand order checked by reading, not run on hardware
//
#include <hip/hip_runtime.h>

typedef __attribute__((ext_vector_type(16))) __bf16 v16bf;
typedef __attribute__((ext_vector_type(8)))  float  v8f;

#define NHEADS 8
#define CCH    256
#define HDIM   32
#define SLEN   256
#define NQ     9
#define HH     64
#define WW     64
#define BATCH  16

// ---------------- workspace layout (bytes) ----------------
// packed conv_w (bf16 A-fragments): 16 mtiles * 8 ksteps * 32 lanes * 16 el * 2B = 131072
// qproj  (f32 [9][256])                                                        =   9216
// o      (f32 [16][9][256])                                                    = 147456
// kern   (f32 [16][256][9])                                                    = 147456
#define WS_PACKED 0
#define WS_QPROJ  131072
#define WS_O      140288
#define WS_KERN   287744

// ---------------------------------------------------------------------------
// Pre-pack conv_w into the CDNA5 WMMA bf16 A-matrix (16x32) per-lane layout:
// lane L holds M = L%16, K set = {b..b+7} U {b+16..b+23} with b=(L/16)*8.
// element e<8 -> K=b+e ; e>=8 -> K=b+8+e.
// ---------------------------------------------------------------------------
__global__ void pack_convw(const float* __restrict__ conv_w, __bf16* __restrict__ packed) {
    int gid  = blockIdx.x * 256 + threadIdx.x;     // 0 .. 65535
    int e    = gid & 15;
    int lane = (gid >> 4) & 31;
    int kk   = (gid >> 9) & 7;
    int mt   = gid >> 12;
    int m    = mt * 16 + (lane & 15);
    int base = (lane >> 4) * 8;
    int K    = kk * 32 + base + (e < 8 ? e : e + 8);
    packed[gid] = (__bf16)conv_w[m * CCH + K];
}

// ---------------------------------------------------------------------------
// q_proj[qq][d] = sum_c w0[qq][c] * Wq[c][d]   (batch independent)
// ---------------------------------------------------------------------------
__global__ void qproj_kernel(const float* __restrict__ w0, const float* __restrict__ Wq,
                             float* __restrict__ qproj) {
    for (int idx = threadIdx.x; idx < NQ * CCH; idx += 256) {
        int qq = idx / CCH, d = idx % CCH;
        float acc = 0.f;
        for (int c = 0; c < CCH; ++c) acc += w0[qq * CCH + c] * Wq[c * CCH + d];
        qproj[idx] = acc;
    }
}

// ---------------------------------------------------------------------------
// Per (batch, head) attention: k/v projections, scores, softmax, o.
// Thread t == sequence position s. k-accumulators stay in registers.
// ---------------------------------------------------------------------------
__global__ void attn_kernel(const float* __restrict__ features, const float* __restrict__ Wk,
                            const float* __restrict__ Wv, const float* __restrict__ qproj,
                            float* __restrict__ o_out) {
    __shared__ float vh[SLEN * HDIM];   // 32 KB
    __shared__ float amat[NQ * SLEN];   // 9 KB
    __shared__ float qh[NQ * HDIM];
    __shared__ float red[256];

    int b  = blockIdx.x >> 3;
    int h  = blockIdx.x & 7;
    int hb = h * HDIM;
    int s  = threadIdx.x;

    for (int idx = s; idx < NQ * HDIM; idx += 256)
        qh[idx] = qproj[(idx / HDIM) * CCH + hb + (idx % HDIM)];

    float ak[HDIM], av[HDIM];
#pragma unroll
    for (int d = 0; d < HDIM; ++d) { ak[d] = 0.f; av[d] = 0.f; }

    const float* frow = features + ((size_t)b * SLEN + s) * CCH;
    for (int c = 0; c < CCH; ++c) {
        float f = frow[c];
        const float* wkr = Wk + c * CCH + hb;
        const float* wvr = Wv + c * CCH + hb;
#pragma unroll
        for (int d = 0; d < HDIM; ++d) { ak[d] += f * wkr[d]; av[d] += f * wvr[d]; }
    }
#pragma unroll
    for (int d = 0; d < HDIM; ++d) vh[s * HDIM + d] = av[d];
    __syncthreads();

    const float scale = 0.17677669529663687f;  // 1/sqrt(32)
    for (int qq = 0; qq < NQ; ++qq) {
        float sc = 0.f;
#pragma unroll
        for (int d = 0; d < HDIM; ++d) sc += qh[qq * HDIM + d] * ak[d];
        sc *= scale;
        red[s] = sc; __syncthreads();
        for (int off = 128; off > 0; off >>= 1) {
            if (s < off) red[s] = fmaxf(red[s], red[s + off]);
            __syncthreads();
        }
        float mx = red[0]; __syncthreads();
        float ex = __expf(sc - mx);
        red[s] = ex; __syncthreads();
        for (int off = 128; off > 0; off >>= 1) {
            if (s < off) red[s] += red[s + off];
            __syncthreads();
        }
        float sm = red[0]; __syncthreads();
        amat[qq * SLEN + s] = ex / sm;
    }
    __syncthreads();

    for (int idx = s; idx < NQ * HDIM; idx += 256) {
        int qq = idx / HDIM, d = idx % HDIM;
        float acc = 0.f;
        for (int ss = 0; ss < SLEN; ++ss) acc += amat[qq * SLEN + ss] * vh[ss * HDIM + d];
        o_out[((size_t)b * NQ + qq) * CCH + hb + d] = acc;
    }
}

// ---------------------------------------------------------------------------
// kern[b][c][qq] = sum_hd o[b][qq][hd] * Wo[hd][c]   (w = o @ Wo, transposed)
// ---------------------------------------------------------------------------
__global__ void mkkern(const float* __restrict__ o_in, const float* __restrict__ Wo,
                       float* __restrict__ kern) {
    int b = blockIdx.x;
    int c = threadIdx.x;
    for (int qq = 0; qq < NQ; ++qq) {
        float acc = 0.f;
        for (int hd = 0; hd < CCH; ++hd)
            acc += o_in[((size_t)b * NQ + qq) * CCH + hd] * Wo[hd * CCH + c];
        kern[((size_t)b * CCH + c) * NQ + qq] = acc;
    }
}

// ---------------------------------------------------------------------------
// Fused dynamic depthwise 3x3 + 1x1 conv.
// Block = (batch, 4-row x 16-col spatial tile). 256 threads = 8 waves.
// Phase 1: thread == channel; sliding-window depthwise; write bf16 y into LDS
//          directly in WMMA B-fragment (32x16) layout.
// Phase 2: each wave owns 32 output channels (2 M-tiles); K-loop over C=256
//          in 8 steps of 32 with v_wmma_f32_16x16x32_bf16 (f32 accumulate).
// ---------------------------------------------------------------------------
__global__ void fused_main(const float* __restrict__ x, const float* __restrict__ kern,
                           const __bf16* __restrict__ packed, const float* __restrict__ conv_b,
                           float* __restrict__ out) {
    __shared__ alignas(32) __bf16 yfrag[8 * 4 * 32 * 16];  // [kstep][ntile][lane][e], 32 KB

    int blk  = blockIdx.x;
    int b    = blk >> 6;
    int tile = blk & 63;
    int th   = tile >> 2, tw = tile & 3;
    int h0   = th * 4, w0 = tw * 16;
    int c    = threadIdx.x;  // channel

    float tp[9];
#pragma unroll
    for (int i = 0; i < 9; ++i) tp[i] = kern[((size_t)b * CCH + c) * NQ + i];
    const float* xc = x + ((size_t)b * CCH + c) * (HH * WW);

    // fragment coords for this channel (constant across the tile)
    int kkc = c >> 5, kc = c & 31;
    int hi  = (kc >> 3) & 1;
    int e   = (kc & 7) | ((kc >> 4) << 3);

    for (int r = 0; r < 4; ++r) {
        int hh = h0 + r;
        bool tv = (hh - 1) >= 0;
        bool bv = (hh + 1) < HH;
        int  cL = w0 - 1;
        bool lv = cL >= 0;
        float a0 = (lv && tv) ? xc[(hh - 1) * WW + cL] : 0.f;
        float a1 = lv ? xc[hh * WW + cL] : 0.f;
        float a2 = (lv && bv) ? xc[(hh + 1) * WW + cL] : 0.f;
        float b0 = tv ? xc[(hh - 1) * WW + w0] : 0.f;
        float b1 = xc[hh * WW + w0];
        float b2 = bv ? xc[(hh + 1) * WW + w0] : 0.f;
#pragma unroll
        for (int col = 0; col < 16; ++col) {
            int  cw = w0 + col + 1;
            bool rv = cw < WW;
            float c0 = (rv && tv) ? xc[(hh - 1) * WW + cw] : 0.f;
            float c1 = rv ? xc[hh * WW + cw] : 0.f;
            float c2 = (rv && bv) ? xc[(hh + 1) * WW + cw] : 0.f;
            float y = tp[0] * a0 + tp[1] * b0 + tp[2] * c0
                    + tp[3] * a1 + tp[4] * b1 + tp[5] * c1
                    + tp[6] * a2 + tp[7] * b2 + tp[8] * c2;
            int lane = col + (hi << 4);
            yfrag[(((kkc << 2) + r) * 32 + lane) * 16 + e] = (__bf16)y;
            a0 = b0; a1 = b1; a2 = b2;
            b0 = c0; b1 = c1; b2 = c2;
        }
    }
    __syncthreads();

    int wv = threadIdx.x >> 5;   // wave id: owns O rows [wv*32, wv*32+32)
    int ln = threadIdx.x & 31;

    v8f acc[2][4];
#pragma unroll
    for (int mi = 0; mi < 2; ++mi)
#pragma unroll
        for (int nt = 0; nt < 4; ++nt)
            acc[mi][nt] = (v8f){0.f, 0.f, 0.f, 0.f, 0.f, 0.f, 0.f, 0.f};

    const v16bf* Ap = (const v16bf*)packed;
#pragma unroll
    for (int kk = 0; kk < 8; ++kk) {
        v16bf A0 = Ap[((wv * 2 + 0) * 8 + kk) * 32 + ln];
        v16bf A1 = Ap[((wv * 2 + 1) * 8 + kk) * 32 + ln];
#pragma unroll
        for (int nt = 0; nt < 4; ++nt) {
            v16bf Bv = *(const v16bf*)&yfrag[(((kk << 2) + nt) * 32 + ln) * 16];
            acc[0][nt] = __builtin_amdgcn_wmma_f32_16x16x32_bf16(
                false, A0, false, Bv, (short)0, acc[0][nt], false, false);
            acc[1][nt] = __builtin_amdgcn_wmma_f32_16x16x32_bf16(
                false, A1, false, Bv, (short)0, acc[1][nt], false, false);
        }
    }

    // C/D layout: VGPR j, lanes 0-15 -> M=j ; lanes 16-31 -> M=j+8 ; N = lane%16
    int colw = ln & 15;
    int half = ln >> 4;
#pragma unroll
    for (int mi = 0; mi < 2; ++mi) {
#pragma unroll
        for (int nt = 0; nt < 4; ++nt) {
#pragma unroll
            for (int j = 0; j < 8; ++j) {
                int m = wv * 32 + mi * 16 + half * 8 + j;
                out[(((size_t)b * CCH + m) * HH + (h0 + nt)) * WW + (w0 + colw)] =
                    acc[mi][nt][j] + conv_b[m];
            }
        }
    }
}

// ---------------------------------------------------------------------------
extern "C" void kernel_launch(void* const* d_in, const int* in_sizes, int n_in,
                              void* d_out, int out_size, void* d_ws, size_t ws_size,
                              hipStream_t stream) {
    (void)in_sizes; (void)n_in; (void)out_size; (void)ws_size;
    const float* x        = (const float*)d_in[0];
    const float* features = (const float*)d_in[1];
    const float* w0       = (const float*)d_in[2];
    const float* Wq       = (const float*)d_in[3];
    const float* Wk       = (const float*)d_in[4];
    const float* Wv       = (const float*)d_in[5];
    const float* Wo       = (const float*)d_in[6];
    const float* conv_w   = (const float*)d_in[7];
    const float* conv_b   = (const float*)d_in[8];
    float* out = (float*)d_out;

    char*   ws     = (char*)d_ws;
    __bf16* packed = (__bf16*)(ws + WS_PACKED);
    float*  qproj  = (float*)(ws + WS_QPROJ);
    float*  obuf   = (float*)(ws + WS_O);
    float*  kernb  = (float*)(ws + WS_KERN);

    pack_convw<<<256, 256, 0, stream>>>(conv_w, packed);
    qproj_kernel<<<1, 256, 0, stream>>>(w0, Wq, qproj);
    attn_kernel<<<BATCH * NHEADS, 256, 0, stream>>>(features, Wk, Wv, qproj, obuf);
    mkkern<<<BATCH, 256, 0, stream>>>(obuf, Wo, kernb);
    fused_main<<<BATCH * 64, 256, 0, stream>>>(x, kernb, packed, conv_b, out);
}
